// MS_ResBlock_71829033058720
// MI455X (gfx1250) — matrix operand order repaired and checked by reading, not compiled, not run on hardware
//
#include <hip/hip_runtime.h>

// ---------------------------------------------------------------------------
// MS-ResBlock (Linear -> BN -> +residual -> LIF) for MI455X / gfx1250.
// x:[16,4096,512] f32, W:[512,512] f32, b,gamma,beta:[512] f32 -> spikes f32.
// GEMM is NT (y[m,n] = sum_k x[m,k]*W[n,k]) via v_wmma_f32_16x16x32_bf16,
// double-buffered LDS staging, BN-stats fused into the epilogue.
// ---------------------------------------------------------------------------

typedef __attribute__((ext_vector_type(16))) __bf16 v16bf;
typedef __attribute__((ext_vector_type(8)))  float  v8f;

#define T_STEPS 16
#define BATCH   4096
#define DIM     512
#define MROWS   (T_STEPS * BATCH)   // 65536
#define BM      128
#define BN      128
#define BK      32

struct __align__(16) F4 { float x, y, z, w; };
struct __align__(16) U4 { unsigned x, y, z, w; };

// pack two f32 into one dword of two bf16 (round-half-up) -- 3 VALU:
// v_add, v_add, v_perm_b32 (bytes {b[3],b[2],a[3],a[2]})
__device__ __forceinline__ unsigned f2bf_pk(float a, float b) {
    unsigned ua = __float_as_uint(a) + 0x8000u;
    unsigned ub = __float_as_uint(b) + 0x8000u;
    return __builtin_amdgcn_perm(ub, ua, 0x07060302u);
}

union AFrag { U4 q[2]; v16bf v; };

// ---------------------------------------------------------------------------
// Kernel 1: y = x @ W^T + b  (bf16 WMMA, f32 accum) + per-feature sum/sumsq
// grid (MROWS/BM, DIM/BN), block 256 (8 waves). No divergence -> EXEC all-1s.
// ---------------------------------------------------------------------------
__global__ __launch_bounds__(256)
void gemm_bn_stats_kernel(const float* __restrict__ x,
                          const float* __restrict__ W,
                          const float* __restrict__ bias,
                          float* __restrict__ y,
                          float* __restrict__ gsum,
                          float* __restrict__ gsumsq) {
    // double-buffered staging: [2] x 128 rows x 16 dwords (bf16 pairs)
    __shared__ __align__(16) unsigned As[2][BM * BK / 2];
    __shared__ __align__(16) unsigned Bs[2][BN * BK / 2];

    const int tid  = threadIdx.x;
    const int wv   = tid >> 5;          // wave 0..7
    const int lane = tid & 31;

    const int mbase = blockIdx.x * BM;
    const int nbase = blockIdx.y * BN;

    // staging: thread -> (row, 16-wide k slab)
    const int srow = tid >> 1;          // 0..127
    const int scol = (tid & 1) * 16;    // 0 or 16 (halves)
    const int sidx = srow * (BK / 2) + (scol >> 1);   // dword index in tile

    const float* xrow = x + (size_t)(mbase + srow) * DIM + scol;
    const float* wrow = W + (size_t)(nbase + srow) * DIM + scol;

    v8f acc[8];
    #pragma unroll
    for (int nt = 0; nt < 8; ++nt)
        #pragma unroll
        for (int i = 0; i < 8; ++i) acc[nt][i] = 0.0f;

    // fragment addressing (per ISA 7.12.2 bf16 layouts, wave32)
    const int arow = (wv << 4) + (lane & 15);     // A: lane -> M row
    const int aoff = (lane >> 4) << 2;            // half-wave K split (dwords)
    const int koff = (lane >> 4) << 2;            // B: half-wave K split (dwords)

    // ---- prologue: stage tile k=0 into buffer 0 ----
    {
        F4 a0 = *(const F4*)(xrow + 0);
        F4 a1 = *(const F4*)(xrow + 4);
        F4 a2 = *(const F4*)(xrow + 8);
        F4 a3 = *(const F4*)(xrow + 12);
        F4 w0 = *(const F4*)(wrow + 0);
        F4 w1 = *(const F4*)(wrow + 4);
        F4 w2 = *(const F4*)(wrow + 8);
        F4 w3 = *(const F4*)(wrow + 12);
        U4* asl = (U4*)&As[0][sidx];
        asl[0] = U4{ f2bf_pk(a0.x, a0.y), f2bf_pk(a0.z, a0.w),
                     f2bf_pk(a1.x, a1.y), f2bf_pk(a1.z, a1.w) };
        asl[1] = U4{ f2bf_pk(a2.x, a2.y), f2bf_pk(a2.z, a2.w),
                     f2bf_pk(a3.x, a3.y), f2bf_pk(a3.z, a3.w) };
        U4* bsl = (U4*)&Bs[0][sidx];
        bsl[0] = U4{ f2bf_pk(w0.x, w0.y), f2bf_pk(w0.z, w0.w),
                     f2bf_pk(w1.x, w1.y), f2bf_pk(w1.z, w1.w) };
        bsl[1] = U4{ f2bf_pk(w2.x, w2.y), f2bf_pk(w2.z, w2.w),
                     f2bf_pk(w3.x, w3.y), f2bf_pk(w3.z, w3.w) };
    }

    int cur = 0;
    for (int k0 = 0; k0 < DIM; k0 += BK) {
        __syncthreads();   // buffer `cur` ready; other buffer free

        const bool has_next = (k0 + BK) < DIM;   // wave-uniform

        // ---- issue next tile's global loads early (overlap with WMMA) ----
        F4 a0, a1, a2, a3, w0, w1, w2, w3;
        if (has_next) {
            const int kn = k0 + BK;
            a0 = *(const F4*)(xrow + kn);
            a1 = *(const F4*)(xrow + kn + 4);
            a2 = *(const F4*)(xrow + kn + 8);
            a3 = *(const F4*)(xrow + kn + 12);
            w0 = *(const F4*)(wrow + kn);
            w1 = *(const F4*)(wrow + kn + 4);
            w2 = *(const F4*)(wrow + kn + 8);
            w3 = *(const F4*)(wrow + kn + 12);
        }

        // ---- fragment loads: dedicated regs per fragment so ds waits pipeline
        const unsigned* Ab = As[cur];
        const unsigned* Bb = Bs[cur];
        AFrag af;
        af.q[0] = *(const U4*)&Ab[arow * 16 + aoff];
        af.q[1] = *(const U4*)&Ab[arow * 16 + aoff + 8];
        AFrag bfr[8];
        #pragma unroll
        for (int nt = 0; nt < 8; ++nt) {
            const int brow = nt * 16 + (lane & 15);   // B: lane -> N col
            bfr[nt].q[0] = *(const U4*)&Bb[brow * 16 + koff];
            bfr[nt].q[1] = *(const U4*)&Bb[brow * 16 + koff + 8];
        }

        #pragma unroll
        for (int nt = 0; nt < 8; ++nt) {
            acc[nt] = __builtin_amdgcn_wmma_f32_16x16x32_bf16(
                false, af.v, false, bfr[nt].v, (short)0, acc[nt], false, false);
        }

        // ---- convert + store next tile into the other buffer ----
        if (has_next) {
            U4* asl = (U4*)&As[cur ^ 1][sidx];
            asl[0] = U4{ f2bf_pk(a0.x, a0.y), f2bf_pk(a0.z, a0.w),
                         f2bf_pk(a1.x, a1.y), f2bf_pk(a1.z, a1.w) };
            asl[1] = U4{ f2bf_pk(a2.x, a2.y), f2bf_pk(a2.z, a2.w),
                         f2bf_pk(a3.x, a3.y), f2bf_pk(a3.z, a3.w) };
            U4* bsl = (U4*)&Bs[cur ^ 1][sidx];
            bsl[0] = U4{ f2bf_pk(w0.x, w0.y), f2bf_pk(w0.z, w0.w),
                         f2bf_pk(w1.x, w1.y), f2bf_pk(w1.z, w1.w) };
            bsl[1] = U4{ f2bf_pk(w2.x, w2.y), f2bf_pk(w2.z, w2.w),
                         f2bf_pk(w3.x, w3.y), f2bf_pk(w3.z, w3.w) };
        }
        cur ^= 1;
    }

    // ---- epilogue: +bias, store y, fold BN partial sums, atomic into table ----
    const int lhalf = lane >> 4;   // C layout: VGPR i -> M = 8*lhalf + i
    const int lcol  = lane & 15;
    #pragma unroll
    for (int nt = 0; nt < 8; ++nt) {
        const int col = nbase + nt * 16 + lcol;
        const float bv = bias[col];
        float ps = 0.0f, ps2 = 0.0f;
        #pragma unroll
        for (int i = 0; i < 8; ++i) {
            float val = acc[nt][i] + bv;
            const int row = mbase + (wv << 4) + (lhalf << 3) + i;
            y[(size_t)row * DIM + col] = val;
            ps  += val;
            ps2 += val * val;
        }
        ps  += __shfl_xor(ps, 16, 32);   // wave32: fold the two half-waves
        ps2 += __shfl_xor(ps2, 16, 32);
        if (lhalf == 0) {
            atomicAdd(&gsum[col],   ps);
            atomicAdd(&gsumsq[col], ps2);
        }
    }
}

// ---------------------------------------------------------------------------
// Kernel 0: zero the stats table (graph-replay safe re-init each launch)
// ---------------------------------------------------------------------------
__global__ void zero_stats_kernel(float* __restrict__ p, int n) {
    int i = blockIdx.x * blockDim.x + threadIdx.x;
    if (i < n) p[i] = 0.0f;
}

// ---------------------------------------------------------------------------
// Kernel 2: mean/var -> (scale, shift) per feature
// ---------------------------------------------------------------------------
__global__ void bn_finalize_kernel(const float* __restrict__ gsum,
                                   const float* __restrict__ gsumsq,
                                   const float* __restrict__ gamma,
                                   const float* __restrict__ beta,
                                   float* __restrict__ scale,
                                   float* __restrict__ shift) {
    const int e = threadIdx.x;   // 512 threads, 1 block
    const float inv_n = 1.0f / (float)MROWS;
    const float m   = gsum[e] * inv_n;
    const float var = gsumsq[e] * inv_n - m * m;
    const float sc  = gamma[e] * __frsqrt_rn(var + 1e-5f);
    scale[e] = sc;
    shift[e] = beta[e] - m * sc;
}

// ---------------------------------------------------------------------------
// Kernel 3: h = y*scale + shift + x ; LIF scan over T (tau=2, v_th=1, hard 0)
// One thread owns FOUR consecutive features of one batch row across all 16
// timesteps -> b128 vmem traffic on x / y / out.
// ---------------------------------------------------------------------------
__global__ __launch_bounds__(256)
void lif_kernel(const float* __restrict__ x,
                const float* __restrict__ y,
                const float* __restrict__ scale,
                const float* __restrict__ shift,
                float* __restrict__ out) {
    const int idx4 = (blockIdx.x * blockDim.x + threadIdx.x) * 4;  // 0..B*D-4
    const int e    = idx4 & (DIM - 1);
    const F4 sc = *(const F4*)(scale + e);
    const F4 sh = *(const F4*)(shift + e);
    float v0 = 0.0f, v1 = 0.0f, v2 = 0.0f, v3 = 0.0f;
    #pragma unroll
    for (int t = 0; t < T_STEPS; ++t) {
        const size_t off = (size_t)t * (BATCH * DIM) + idx4;
        const F4 yv = *(const F4*)(y + off);
        const F4 xv = *(const F4*)(x + off);
        F4 s;
        {
            const float h = fmaf(yv.x, sc.x, sh.x) + xv.x;
            v0 = fmaf(h - v0, 0.5f, v0);
            s.x = (v0 >= 1.0f) ? 1.0f : 0.0f;
            v0 *= (1.0f - s.x);
        }
        {
            const float h = fmaf(yv.y, sc.y, sh.y) + xv.y;
            v1 = fmaf(h - v1, 0.5f, v1);
            s.y = (v1 >= 1.0f) ? 1.0f : 0.0f;
            v1 *= (1.0f - s.y);
        }
        {
            const float h = fmaf(yv.z, sc.z, sh.z) + xv.z;
            v2 = fmaf(h - v2, 0.5f, v2);
            s.z = (v2 >= 1.0f) ? 1.0f : 0.0f;
            v2 *= (1.0f - s.z);
        }
        {
            const float h = fmaf(yv.w, sc.w, sh.w) + xv.w;
            v3 = fmaf(h - v3, 0.5f, v3);
            s.w = (v3 >= 1.0f) ? 1.0f : 0.0f;
            v3 *= (1.0f - s.w);
        }
        *(F4*)(out + off) = s;
    }
}

// ---------------------------------------------------------------------------
extern "C" void kernel_launch(void* const* d_in, const int* in_sizes, int n_in,
                              void* d_out, int out_size, void* d_ws, size_t ws_size,
                              hipStream_t stream) {
    (void)in_sizes; (void)n_in; (void)out_size; (void)ws_size;

    const float* x     = (const float*)d_in[0];   // [16,4096,512]
    const float* W     = (const float*)d_in[1];   // [512,512]
    const float* bias  = (const float*)d_in[2];   // [512]
    const float* gamma = (const float*)d_in[3];   // [512]
    const float* beta  = (const float*)d_in[4];   // [512]
    float*       out   = (float*)d_out;

    // workspace layout: y [65536x512 f32] | sum[512] | sumsq[512] | scale[512] | shift[512]
    float* y      = (float*)d_ws;
    float* stats  = (float*)((char*)d_ws + (size_t)MROWS * DIM * sizeof(float));
    float* gsum   = stats;
    float* gsumsq = stats + 512;
    float* scale  = stats + 1024;
    float* shift  = stats + 1536;

    zero_stats_kernel<<<4, 256, 0, stream>>>(stats, 1024);

    dim3 ggrid(MROWS / BM, DIM / BN);   // (512, 4)
    gemm_bn_stats_kernel<<<ggrid, 256, 0, stream>>>(x, W, bias, y, gsum, gsumsq);

    bn_finalize_kernel<<<1, DIM, 0, stream>>>(gsum, gsumsq, gamma, beta, scale, shift);

    lif_kernel<<<(BATCH * DIM) / (256 * 4), 256, 0, stream>>>(x, y, scale, shift, out);
}
